// OCRAttention_39986145526219
// MI455X (gfx1250) — matrix operand order repaired and checked by reading, not compiled, not run on hardware
//
#include <hip/hip_runtime.h>
#include <hip/hip_bf16.h>
#include <stdint.h>

typedef __attribute__((ext_vector_type(16))) __bf16 v16bf;
typedef __attribute__((ext_vector_type(8)))  __bf16 v8bf;
typedef __attribute__((ext_vector_type(8)))  float  v8f;

#define DMODEL 1024
#define SEQ    2048
#define NHEAD  16
#define HDIM   64
#define MROWS  8192   // B*S

static __device__ __forceinline__ v16bf combine8(v8bf lo, v8bf hi) {
  v16bf r;
#pragma unroll
  for (int i = 0; i < 8; ++i) { r[i] = lo[i]; r[i + 8] = hi[i]; }
  return r;
}

// gfx1250 async global->LDS copy (ASYNCcnt-tracked). lds_off is the 32-bit
// wave-relative LDS byte offset (low 32 bits of a generic pointer to LDS);
// INST_OFFSET is applied to both the LDS and global side per the ISA.
static __device__ __forceinline__ void async_b128(uint32_t lds_off,
                                                  uint64_t gaddr, int ioff) {
  asm volatile("global_load_async_to_lds_b128 %0, %1, off offset:%2"
               :: "v"(lds_off), "v"(gaddr), "i"(ioff) : "memory");
}
static __device__ __forceinline__ void wait_async0() {
  asm volatile("s_wait_asynccnt 0" ::: "memory");
}

// ---------------------------------------------------------------- convert x
__global__ void cvt_f32_to_bf16(const float* __restrict__ in,
                                __bf16* __restrict__ out, int n) {
  int i = (blockIdx.x * blockDim.x + threadIdx.x) * 4;
  if (i + 3 < n) {
    float4 v = *(const float4*)(in + i);
    out[i + 0] = (__bf16)v.x; out[i + 1] = (__bf16)v.y;
    out[i + 2] = (__bf16)v.z; out[i + 3] = (__bf16)v.w;
  }
}

// ------------------------------------------------- transpose weight -> bf16
__global__ void transpose_w_bf16(const float* __restrict__ in,
                                 __bf16* __restrict__ out) {
  int i = blockIdx.x * blockDim.x + threadIdx.x;   // 0 .. 1024*1024-1
  int k = i >> 10, n = i & 1023;
  out[(size_t)n * DMODEL + k] = (__bf16)in[(size_t)k * DMODEL + n];
}

// ------------------------------------------------------------- WMMA GEMM
// C[M,1024] = A[M,1024] @ B, B given transposed as BT[n][k] (bf16).
// WG = 128 threads / 4 waves; WG tile 128x64; wave strip 32x64.
// Double-buffered LDS, staged with async global->LDS copies.
__global__ __launch_bounds__(128) void gemm_bf16_1024(
    const __bf16* __restrict__ A, const __bf16* __restrict__ BT,
    const float* __restrict__ bias, const float* __restrict__ bias2,
    __bf16* __restrict__ outb, float* __restrict__ outf) {
  __shared__ __attribute__((aligned(32))) __bf16 As[2][128][64];  // 32 KB
  __shared__ __attribute__((aligned(32))) __bf16 Bs[2][64][64];   // 16 KB

  const int tid  = threadIdx.x;
  const int wave = tid >> 5, lane = tid & 31;
  const int hh = lane >> 4, lm = lane & 15;
  const int tn = blockIdx.x & 15, tm = blockIdx.x >> 4;  // grid 64*16
  const int m0 = tm * 128, n0 = tn * 64;
  const int br = tid >> 1, bc = (tid & 1) * 32;

  v8f acc[2][4] = {};

  auto stage = [&](int buf, int k0) {
    const uint64_t ag =
        (uint64_t)(uintptr_t)(A + (size_t)(m0 + tid) * DMODEL + k0);
    const uint32_t al = (uint32_t)(uintptr_t)&As[buf][tid][0];
#pragma unroll
    for (int i = 0; i < 8; ++i) async_b128(al, ag, 16 * i);
    const uint64_t bg =
        (uint64_t)(uintptr_t)(BT + (size_t)(n0 + br) * DMODEL + k0 + bc);
    const uint32_t bl = (uint32_t)(uintptr_t)&Bs[buf][br][bc];
#pragma unroll
    for (int i = 0; i < 4; ++i) async_b128(bl, bg, 16 * i);
  };

  stage(0, 0);
  for (int it = 0; it < DMODEL / 64; ++it) {
    const int cur = it & 1;
    wait_async0();      // my async copies for buf `cur` have landed
    __syncthreads();    // everyone's have; prior reads of !cur done
    if (it + 1 < DMODEL / 64) stage(cur ^ 1, (it + 1) * 64);

#pragma unroll
    for (int kc = 0; kc < 2; ++kc) {
      v16bf af[2];
#pragma unroll
      for (int mt = 0; mt < 2; ++mt) {
        const __bf16* ar = &As[cur][wave * 32 + mt * 16 + lm][kc * 32];
        af[mt] = combine8(*(const v8bf*)(ar + 8 * hh),
                          *(const v8bf*)(ar + 16 + 8 * hh));
      }
#pragma unroll
      for (int nt = 0; nt < 4; ++nt) {
        v16bf bfrag = combine8(*(const v8bf*)(&Bs[cur][nt * 16 + lm][kc * 32 + 8 * hh]),
                               *(const v8bf*)(&Bs[cur][nt * 16 + lm][kc * 32 + 16 + 8 * hh]));
#pragma unroll
        for (int mt = 0; mt < 2; ++mt)
          acc[mt][nt] = __builtin_amdgcn_wmma_f32_16x16x32_bf16(
              false, af[mt], false, bfrag, (short)0, acc[mt][nt], false, false);
      }
    }
  }

#pragma unroll
  for (int mt = 0; mt < 2; ++mt)
#pragma unroll
    for (int nt = 0; nt < 4; ++nt) {
      const int ng = n0 + nt * 16 + lm;
      float bsv = bias[ng] + (bias2 ? bias2[ng] : 0.0f);
#pragma unroll
      for (int j = 0; j < 8; ++j) {
        const int mg = m0 + wave * 32 + mt * 16 + j + 8 * hh;
        float val = acc[mt][nt][j] + bsv;
        if (outb) outb[(size_t)mg * DMODEL + ng] = (__bf16)val;
        else      outf[(size_t)mg * DMODEL + ng] = val;
      }
    }
}

// --------------------------------------------------- flash attention (bf16)
// Q,K,V : bf16 [8192][1024] (row = b*S+s, col = h*64+d). O same layout.
__global__ __launch_bounds__(128) void flash_attn_band(
    const __bf16* __restrict__ Q, const __bf16* __restrict__ K,
    const __bf16* __restrict__ V, __bf16* __restrict__ O) {
  __shared__ __attribute__((aligned(32))) __bf16 Qs[64][64];
  __shared__ __attribute__((aligned(32))) __bf16 Ks[64][64];
  __shared__ __attribute__((aligned(32))) __bf16 Vts[64][64];  // [dim][key]
  __shared__ __attribute__((aligned(32))) __bf16 Ps[4][16][64];

  const int tid  = threadIdx.x;
  const int wave = tid >> 5, lane = tid & 31;
  const int hh = lane >> 4, lm = lane & 15;
  const int qt = blockIdx.x & 31;
  const int h  = (blockIdx.x >> 5) & 15;
  const int b  = blockIdx.x >> 9;
  const int q0 = qt * 64;
  const size_t baseRow = (size_t)b * SEQ;
  const int hc = h * HDIM;
  const int r = tid >> 1, c0 = (tid & 1) * 32;

  {  // stage Q block once (async DMA into LDS)
    const uint64_t qg =
        (uint64_t)(uintptr_t)(Q + (baseRow + q0 + r) * DMODEL + hc + c0);
    const uint32_t ql = (uint32_t)(uintptr_t)&Qs[r][c0];
#pragma unroll
    for (int i = 0; i < 4; ++i) async_b128(ql, qg, 16 * i);
  }
  wait_async0();
  __syncthreads();

  v16bf qf[2];
#pragma unroll
  for (int c = 0; c < 2; ++c) {
    const __bf16* qr = &Qs[wave * 16 + lm][32 * c];
    qf[c] = combine8(*(const v8bf*)(qr + 8 * hh),
                     *(const v8bf*)(qr + 16 + 8 * hh));
  }

  v8f o[4] = {};
  float mrow[8], lrow[8];
#pragma unroll
  for (int j = 0; j < 8; ++j) { mrow[j] = -1e30f; lrow[j] = 0.0f; }

  for (int kt = 0; kt < SEQ / 64; ++kt) {
    __syncthreads();  // previous-iteration LDS reads complete
    {   // stage K (async, natural) and V (transposed via VGPRs)
      const size_t krow = baseRow + kt * 64 + r;
      const uint64_t kg = (uint64_t)(uintptr_t)(K + krow * DMODEL + hc + c0);
      const uint32_t kl = (uint32_t)(uintptr_t)&Ks[r][c0];
#pragma unroll
      for (int i = 0; i < 4; ++i) async_b128(kl, kg, 16 * i);

      v16bf v0 = *(const v16bf*)(V + krow * DMODEL + hc + c0);
      v16bf v1 = *(const v16bf*)(V + krow * DMODEL + hc + c0 + 16);
#pragma unroll
      for (int c = 0; c < 16; ++c) {
        Vts[c0 + c][r]      = v0[c];
        Vts[c0 + 16 + c][r] = v1[c];
      }
      if (kt + 1 < SEQ / 64)
        __builtin_prefetch(V + (krow + 64) * DMODEL + hc + c0, 0, 3);
    }
    wait_async0();
    __syncthreads();

    // ---- scores: 16x64 per wave, Q @ K^T
    v8f s[4] = {};
#pragma unroll
    for (int nt = 0; nt < 4; ++nt)
#pragma unroll
      for (int c = 0; c < 2; ++c) {
        v16bf kf = combine8(*(const v8bf*)(&Ks[nt * 16 + lm][32 * c + 8 * hh]),
                            *(const v8bf*)(&Ks[nt * 16 + lm][32 * c + 16 + 8 * hh]));
        s[nt] = __builtin_amdgcn_wmma_f32_16x16x32_bf16(
            false, qf[c], false, kf, (short)0, s[nt], false, false);
      }

    // ---- scale + banded bias
#pragma unroll
    for (int nt = 0; nt < 4; ++nt) {
      const int kg = kt * 64 + nt * 16 + lm;
#pragma unroll
      for (int j = 0; j < 8; ++j) {
        const int qg = q0 + wave * 16 + j + 8 * hh;
        float sv = s[nt][j] * 0.125f;
        int d = qg - kg; d = d < 0 ? -d : d;
        s[nt][j] = sv + ((d <= 2) ? 0.1f : 0.0f);
      }
    }

    // ---- online softmax over this key tile
#pragma unroll
    for (int j = 0; j < 8; ++j) {
      float mx = fmaxf(fmaxf(s[0][j], s[1][j]), fmaxf(s[2][j], s[3][j]));
#pragma unroll
      for (int off = 1; off < 16; off <<= 1)
        mx = fmaxf(mx, __shfl_xor(mx, off, 32));
      float mnew = fmaxf(mrow[j], mx);
      float corr = __expf(mrow[j] - mnew);
      mrow[j] = mnew;
      float rs = 0.0f;
#pragma unroll
      for (int nt = 0; nt < 4; ++nt) {
        float p = __expf(s[nt][j] - mnew);
        s[nt][j] = p;
        rs += p;
      }
#pragma unroll
      for (int off = 1; off < 16; off <<= 1)
        rs += __shfl_xor(rs, off, 32);
      lrow[j] = lrow[j] * corr + rs;
#pragma unroll
      for (int nt = 0; nt < 4; ++nt) o[nt][j] *= corr;
    }

    // ---- P (f32 C-layout) -> LDS bf16 in A-layout order
#pragma unroll
    for (int nt = 0; nt < 4; ++nt)
#pragma unroll
      for (int j = 0; j < 8; ++j)
        Ps[wave][j + 8 * hh][nt * 16 + lm] = (__bf16)s[nt][j];
    __syncthreads();

    // ---- O += P @ V
#pragma unroll
    for (int c = 0; c < 2; ++c) {
      const __bf16* pr = &Ps[wave][lm][32 * c];
      v16bf pf = combine8(*(const v8bf*)(pr + 8 * hh),
                          *(const v8bf*)(pr + 16 + 8 * hh));
#pragma unroll
      for (int nt = 0; nt < 4; ++nt) {
        v16bf vf = combine8(*(const v8bf*)(&Vts[nt * 16 + lm][32 * c + 8 * hh]),
                            *(const v8bf*)(&Vts[nt * 16 + lm][32 * c + 16 + 8 * hh]));
        o[nt] = __builtin_amdgcn_wmma_f32_16x16x32_bf16(
            false, pf, false, vf, (short)0, o[nt], false, false);
      }
    }
  }

  // ---- normalize + store
#pragma unroll
  for (int nt = 0; nt < 4; ++nt)
#pragma unroll
    for (int j = 0; j < 8; ++j) {
      const int qg = q0 + wave * 16 + j + 8 * hh;
      float val = o[nt][j] / lrow[j];
      O[(baseRow + qg) * DMODEL + hc + nt * 16 + lm] = (__bf16)val;
    }
}

// ---------------------------------------------------------------- launcher
extern "C" void kernel_launch(void* const* d_in, const int* in_sizes, int n_in,
                              void* d_out, int out_size, void* d_ws,
                              size_t ws_size, hipStream_t stream) {
  const float* x        = (const float*)d_in[0];
  // d_in[1] = ocr_context (shape-only, unused)
  const float* Wq       = (const float*)d_in[2];
  const float* bq       = (const float*)d_in[3];
  const float* Wk       = (const float*)d_in[4];
  const float* bk       = (const float*)d_in[5];
  const float* Wv       = (const float*)d_in[6];
  const float* bv       = (const float*)d_in[7];
  const float* Wo       = (const float*)d_in[8];
  const float* bo       = (const float*)d_in[9];
  const float* ocr_bias = (const float*)d_in[10];
  float* out            = (float*)d_out;

  char* ws = (char*)d_ws;
  size_t off = 0;
  auto take = [&](size_t bytes) {
    char* p = ws + off;
    off += (bytes + 255) & ~(size_t)255;
    return p;
  };
  __bf16* xbf = (__bf16*)take((size_t)MROWS * DMODEL * 2);
  __bf16* WqT = (__bf16*)take((size_t)DMODEL * DMODEL * 2);
  __bf16* WkT = (__bf16*)take((size_t)DMODEL * DMODEL * 2);
  __bf16* WvT = (__bf16*)take((size_t)DMODEL * DMODEL * 2);
  __bf16* WoT = (__bf16*)take((size_t)DMODEL * DMODEL * 2);
  __bf16* Qb  = (__bf16*)take((size_t)MROWS * DMODEL * 2);
  __bf16* Kb  = (__bf16*)take((size_t)MROWS * DMODEL * 2);
  __bf16* Vb  = (__bf16*)take((size_t)MROWS * DMODEL * 2);
  __bf16* Ob  = (__bf16*)take((size_t)MROWS * DMODEL * 2);

  // 1. precision conversion / weight transpose
  cvt_f32_to_bf16<<<(MROWS * DMODEL) / (256 * 4), 256, 0, stream>>>(
      x, xbf, MROWS * DMODEL);
  transpose_w_bf16<<<(DMODEL * DMODEL) / 256, 256, 0, stream>>>(Wq, WqT);
  transpose_w_bf16<<<(DMODEL * DMODEL) / 256, 256, 0, stream>>>(Wk, WkT);
  transpose_w_bf16<<<(DMODEL * DMODEL) / 256, 256, 0, stream>>>(Wv, WvT);
  transpose_w_bf16<<<(DMODEL * DMODEL) / 256, 256, 0, stream>>>(Wo, WoT);

  // 2. QKV projections (bias fused; ocr_bias fused into Q)
  const int gemmGrid = (MROWS / 128) * (DMODEL / 64);  // 1024
  gemm_bf16_1024<<<gemmGrid, 128, 0, stream>>>(xbf, WqT, bq, ocr_bias, Qb, nullptr);
  gemm_bf16_1024<<<gemmGrid, 128, 0, stream>>>(xbf, WkT, bk, nullptr,  Kb, nullptr);
  gemm_bf16_1024<<<gemmGrid, 128, 0, stream>>>(xbf, WvT, bv, nullptr,  Vb, nullptr);

  // 3. flash attention with banded bias: 4 batches * 16 heads * 32 q-tiles
  flash_attn_band<<<4 * 16 * (SEQ / 64), 128, 0, stream>>>(Qb, Kb, Vb, Ob);

  // 4. output projection -> f32 d_out
  gemm_bf16_1024<<<gemmGrid, 128, 0, stream>>>(Ob, WoT, bo, nullptr, nullptr, out);
}
